// DecoderLayer_33758442946809
// MI455X (gfx1250) — matrix operand order repaired and checked by reading, not compiled, not run on hardware
//
#include <hip/hip_runtime.h>
#include <hip/hip_bf16.h>
#include <math.h>

// ---------------------------------------------------------------------------
// Decoder layer for MI455X (gfx1250, wave32, WMMA).
// All GEMMs run through v_wmma_f32_16x16x32_bf16 (bf16 in, fp32 accumulate).
// cat2 @ W2 is algebraically reduced to attn2 @ sum_h(W2 blocks).
// Attention scores are materialized per head (fits L2), softmax'ed, then PV.
// GEMM: 128x128x32 tile, 8 waves, double-buffered LDS, one barrier/iter.
// ---------------------------------------------------------------------------

#define B_  2
#define S_  2048
#define D_  1024
#define H_  16
#define DF_ 4096
#define DK_ 64
#define BS_ (B_*S_)

typedef unsigned short u16;
typedef __attribute__((ext_vector_type(16))) __bf16        bf16x16;
typedef __attribute__((ext_vector_type(8)))  float         f32x8;
typedef __attribute__((ext_vector_type(4)))  unsigned int  u32x4;
typedef __attribute__((ext_vector_type(8)))  unsigned short u16x8;

__device__ __forceinline__ u16 f2bf(float f) {
  union { float f; unsigned u; } x; x.f = f;
  unsigned u = x.u;
  u += 0x7fffu + ((u >> 16) & 1u);   // round-to-nearest-even
  return (u16)(u >> 16);
}

__device__ __forceinline__ float waveMax(float v) {
  #pragma unroll
  for (int o = 16; o > 0; o >>= 1) v = fmaxf(v, __shfl_xor(v, o, 32));
  return v;
}
__device__ __forceinline__ float waveSum(float v) {
  #pragma unroll
  for (int o = 16; o > 0; o >>= 1) v += __shfl_xor(v, o, 32);
  return v;
}

// ---------------------------------------------------------------------------
// Generic bf16 WMMA GEMM.
//   TRANSB=0: C[M,N] = A[M,K] * B[K,N]        (B row-major)
//   TRANSB=1: C[M,N] = A[M,K] * B2[N,K]^T     (B2 row-major -> no transpose mat.)
//   FULL=1  : M,N are multiples of 128 -> no bounds guards on staging/epilogue.
// Epilogue: v = acc*alpha + bias[col]; optional ReLU; write fp32 (Cf) or
// bf16 (Cbf). Batched via gridDim.z with element strides.
// Block: 256 threads = 8 waves; tile 128x128x32; wave tile 32x64 (2x4 WMMA).
// Double-buffered LDS (2 x (8KB A + 8KB B) padded): one barrier per k-step.
// ---------------------------------------------------------------------------
template<int TRANSB, int FULL>
__global__ __launch_bounds__(256) void gemm_bf16_wmma(
    const u16* __restrict__ A,  long long strideA, int lda,
    const u16* __restrict__ Bm, long long strideB, int ldb,
    float* __restrict__ Cf, u16* __restrict__ Cbf, long long strideC, int ldc,
    int M, int N, int K, float alpha, const float* __restrict__ bias, int relu)
{
  __shared__ __align__(16) u16 shA[2][128 * 40];
  __shared__ __align__(16) u16 shB[2][128 * 40];   // [n][k] layout

  const int tid  = threadIdx.x;
  const int wave = tid >> 5, lane = tid & 31;
  const int wm = wave >> 1, wn = wave & 1;          // 4x2 wave grid
  const int half = lane >> 4, l16 = lane & 15;
  const int bm = blockIdx.y * 128, bn = blockIdx.x * 128;

  A  += (long long)blockIdx.z * strideA;
  Bm += (long long)blockIdx.z * strideB;
  const long long cOff = (long long)blockIdx.z * strideC;

  // ---- per-thread staging coordinates (k0 advances by 32 each step) ----
  // A tile (and B tile when TRANSB): thread covers rows r0 and r0+64, 16B chunk.
  const int rA  = tid >> 2;             // 0..63
  const int kcA = (tid & 3) << 3;       // 0,8,16,24
  // B tile (NN): thread covers k-rows kk0 and kk0+16, 8 cols starting ncB.
  const int kkB = tid >> 4;             // 0..15
  const int ncB = (tid & 15) << 3;      // 0..120

  const f32x8 fzero = {0.f,0.f,0.f,0.f,0.f,0.f,0.f,0.f};
  f32x8 acc[2][4];
  #pragma unroll
  for (int i = 0; i < 2; ++i)
    #pragma unroll
    for (int j = 0; j < 4; ++j) acc[i][j] = fzero;

  const int nk = K >> 5;

  // ---- staging helpers (register prefetch) ----
  u32x4 pa[2], pb4[2];
  u16x8 pb8[2];

  auto fetch = [&](int k0) {
    #pragma unroll
    for (int it = 0; it < 2; ++it) {
      int row = rA + it * 64;
      u32x4 v = {0u,0u,0u,0u};
      if (FULL || (bm + row < M))
        v = *(const u32x4*)(A + (size_t)(bm + row) * lda + k0 + kcA);
      pa[it] = v;
    }
    if (TRANSB) {
      #pragma unroll
      for (int it = 0; it < 2; ++it) {
        int n = rA + it * 64;
        u32x4 v = {0u,0u,0u,0u};
        if (FULL || (bn + n < N))
          v = *(const u32x4*)(Bm + (size_t)(bn + n) * ldb + k0 + kcA);
        pb4[it] = v;
      }
    } else {
      #pragma unroll
      for (int it = 0; it < 2; ++it) {
        int kk = kkB + it * 16;
        u16x8 v = {0,0,0,0,0,0,0,0};
        if (FULL || (bn + ncB < N))
          v = *(const u16x8*)(Bm + (size_t)(k0 + kk) * ldb + bn + ncB);
        pb8[it] = v;
      }
    }
  };

  auto commit = [&](int buf) {
    #pragma unroll
    for (int it = 0; it < 2; ++it)
      *(u32x4*)(shA[buf] + (rA + it * 64) * 40 + kcA) = pa[it];
    if (TRANSB) {
      #pragma unroll
      for (int it = 0; it < 2; ++it)
        *(u32x4*)(shB[buf] + (rA + it * 64) * 40 + kcA) = pb4[it];
    } else {
      #pragma unroll
      for (int it = 0; it < 2; ++it) {
        int kk = kkB + it * 16;
        #pragma unroll
        for (int j = 0; j < 8; ++j) shB[buf][(ncB + j) * 40 + kk] = pb8[it][j];
      }
    }
  };

  // prologue: stage k-block 0 into buffer 0
  fetch(0);
  commit(0);
  __syncthreads();

  for (int kb = 0; kb < nk; ++kb) {
    const int cur = kb & 1;
    const bool more = (kb + 1) < nk;
    if (more) fetch((kb + 1) << 5);     // global prefetch overlaps WMMA below

    // ---- fragments per ISA layouts ----
    union Frag { u32x4 q[2]; bf16x16 v; };
    Frag fa[2], fb[4];
    #pragma unroll
    for (int mt = 0; mt < 2; ++mt) {
      int r = wm * 32 + mt * 16 + l16;
      fa[mt].q[0] = *(const u32x4*)(shA[cur] + r * 40 + half * 8);       // K=half*8+[0..7]
      fa[mt].q[1] = *(const u32x4*)(shA[cur] + r * 40 + 16 + half * 8);  // K=16+half*8+[0..7]
    }
    #pragma unroll
    for (int nt = 0; nt < 4; ++nt) {
      int n = wn * 64 + nt * 16 + l16;
      fb[nt].q[0] = *(const u32x4*)(shB[cur] + n * 40 + half * 8);       // K=half*8+[0..7]
      fb[nt].q[1] = *(const u32x4*)(shB[cur] + n * 40 + 16 + half * 8);  // K=16+half*8+[0..7]
    }
    #pragma unroll
    for (int mt = 0; mt < 2; ++mt)
      #pragma unroll
      for (int nt = 0; nt < 4; ++nt)
        acc[mt][nt] = __builtin_amdgcn_wmma_f32_16x16x32_bf16(
            false, fa[mt].v, false, fb[nt].v, (short)0, acc[mt][nt],
            false, false);

    if (more) {
      commit(cur ^ 1);    // safe: everyone left buf cur^1 at previous barrier
      __syncthreads();
    }
  }

  // ---- epilogue: C/D layout VGPR r -> M = r + 8*half, col = l16 ----
  #pragma unroll
  for (int mt = 0; mt < 2; ++mt)
    #pragma unroll
    for (int nt = 0; nt < 4; ++nt) {
      int col = bn + wn * 64 + nt * 16 + l16;
      if (!FULL && col >= N) continue;
      float bv = bias ? bias[col] : 0.f;
      #pragma unroll
      for (int r = 0; r < 8; ++r) {
        int row = bm + wm * 32 + mt * 16 + half * 8 + r;
        if (!FULL && row >= M) continue;
        float v = acc[mt][nt][r] * alpha + bv;
        if (relu) v = v > 0.f ? v : 0.f;
        long long idx = cOff + (long long)row * ldc + col;
        if (Cbf) Cbf[idx] = f2bf(v);
        else     Cf[idx]  = v;
      }
    }
}

// ---------------------------------------------------------------------------
// Row softmax over length S_, optional causal mask (col <= row%S_ valid).
// Writes bf16 probabilities (zeros beyond the causal limit).
// ---------------------------------------------------------------------------
__global__ __launch_bounds__(256) void k_softmax_rows(
    const float* __restrict__ sc, u16* __restrict__ p, int causal)
{
  __shared__ float red[8];
  const int row = blockIdx.x;
  const int s = row % S_;
  const int limit = causal ? (s + 1) : S_;
  const float* r = sc + (size_t)row * S_;
  u16* o = p + (size_t)row * S_;
  const int tid = threadIdx.x;

  float mx = -1e30f;
  for (int t = tid; t < limit; t += 256) mx = fmaxf(mx, r[t]);
  mx = waveMax(mx);
  if ((tid & 31) == 0) red[tid >> 5] = mx;
  __syncthreads();
  float m = red[0];
  #pragma unroll
  for (int i = 1; i < 8; ++i) m = fmaxf(m, red[i]);
  __syncthreads();

  float sum = 0.f;
  for (int t = tid; t < limit; t += 256) sum += __expf(r[t] - m);
  sum = waveSum(sum);
  if ((tid & 31) == 0) red[tid >> 5] = sum;
  __syncthreads();
  float tot = 0.f;
  #pragma unroll
  for (int i = 0; i < 8; ++i) tot += red[i];
  const float inv = 1.f / tot;

  for (int t = tid; t < S_; t += 256)
    o[t] = f2bf(t < limit ? __expf(r[t] - m) * inv : 0.f);
}

// ---------------------------------------------------------------------------
// out = LayerNorm(gemm + y + bias) * g + b ; writes bf16 and/or fp32.
// One block per row of length D_ (1024) -> 4 elements/thread.
// ---------------------------------------------------------------------------
__global__ __launch_bounds__(256) void k_res_bias_ln(
    const float* __restrict__ gf, const float* __restrict__ y,
    const float* __restrict__ bias, const float* __restrict__ g,
    const float* __restrict__ b, u16* __restrict__ obf,
    float* __restrict__ of)
{
  __shared__ float red[16];
  const int row = blockIdx.x, tid = threadIdx.x;
  const size_t base = (size_t)row * D_;
  float vals[4], s = 0.f, s2 = 0.f;
  #pragma unroll
  for (int i = 0; i < 4; ++i) {
    int c = tid + i * 256;
    float v = gf[base + c] + y[base + c] + bias[c];
    vals[i] = v; s += v; s2 += v * v;
  }
  s = waveSum(s); s2 = waveSum(s2);
  if ((tid & 31) == 0) { red[tid >> 5] = s; red[8 + (tid >> 5)] = s2; }
  __syncthreads();
  float S1 = 0.f, S2 = 0.f;
  #pragma unroll
  for (int i = 0; i < 8; ++i) { S1 += red[i]; S2 += red[8 + i]; }
  const float mean = S1 * (1.f / D_);
  const float var  = S2 * (1.f / D_) - mean * mean;
  const float rs   = rsqrtf(var + 1e-5f);
  #pragma unroll
  for (int i = 0; i < 4; ++i) {
    int c = tid + i * 256;
    float o = (vals[i] - mean) * rs * g[c] + b[c];
    if (obf) obf[base + c] = f2bf(o);
    if (of)  of[base + c]  = o;
  }
}

// ---------------------------------------------------------------------------
// Small prep kernels.
// ---------------------------------------------------------------------------
__global__ void k_cvt_bf16(const float* __restrict__ in, u16* __restrict__ out, int n) {
  int i = blockIdx.x * 256 + threadIdx.x;
  if (i < n) out[i] = f2bf(in[i]);
}

// Wq/Wk/Wv [H,D,DK] -> packed [D, 3*D] bf16 (col = which*D + h*DK + kk)
__global__ void k_pack_qkvw(const float* __restrict__ Wq, const float* __restrict__ Wk,
                            const float* __restrict__ Wv, u16* __restrict__ out) {
  int i = blockIdx.x * 256 + threadIdx.x;
  if (i >= D_ * 3 * D_) return;
  int d = i / (3 * D_), c = i % (3 * D_);
  int which = c / D_, hh = (c % D_) / DK_, kk = c % DK_;
  const float* W = (which == 0) ? Wq : (which == 1) ? Wk : Wv;
  out[i] = f2bf(W[((size_t)hh * D_ + d) * DK_ + kk]);
}

__global__ void k_pack_qkvb(const float* __restrict__ bq, const float* __restrict__ bk,
                            const float* __restrict__ bv, float* __restrict__ out) {
  int c = blockIdx.x * 256 + threadIdx.x;
  if (c >= 3 * D_) return;
  int which = c / D_, hh = (c % D_) / DK_, kk = c % DK_;
  const float* bb = (which == 0) ? bq : (which == 1) ? bk : bv;
  out[c] = bb[hh * DK_ + kk];
}

// W2 [H*D, D] -> sum over the H row-blocks -> [D, D] bf16 (tile(attn2,h) @ W2)
__global__ void k_sum_w2(const float* __restrict__ W2, u16* __restrict__ out) {
  int i = blockIdx.x * 256 + threadIdx.x;
  if (i >= D_ * D_) return;
  int d = i / D_, e = i % D_;
  float s = 0.f;
  #pragma unroll
  for (int hh = 0; hh < H_; ++hh) s += W2[((size_t)hh * D_ + d) * D_ + e];
  out[i] = f2bf(s);
}

// ---------------------------------------------------------------------------
// Host orchestration.
// ---------------------------------------------------------------------------
static void launch_gemm(int transB,
                        const u16* A, long long sA, int lda,
                        const u16* Bm, long long sB, int ldb,
                        float* Cf, u16* Cbf, long long sC, int ldc,
                        int M, int N, int K, float alpha, const float* bias,
                        int relu, int batch, hipStream_t st)
{
  dim3 grid((N + 127) / 128, (M + 127) / 128, batch), block(256);
  const bool full = (M % 128 == 0) && (N % 128 == 0);
  if (transB) {
    if (full) gemm_bf16_wmma<1,1><<<grid, block, 0, st>>>(A, sA, lda, Bm, sB, ldb,
                    Cf, Cbf, sC, ldc, M, N, K, alpha, bias, relu);
    else      gemm_bf16_wmma<1,0><<<grid, block, 0, st>>>(A, sA, lda, Bm, sB, ldb,
                    Cf, Cbf, sC, ldc, M, N, K, alpha, bias, relu);
  } else {
    if (full) gemm_bf16_wmma<0,1><<<grid, block, 0, st>>>(A, sA, lda, Bm, sB, ldb,
                    Cf, Cbf, sC, ldc, M, N, K, alpha, bias, relu);
    else      gemm_bf16_wmma<0,0><<<grid, block, 0, st>>>(A, sA, lda, Bm, sB, ldb,
                    Cf, Cbf, sC, ldc, M, N, K, alpha, bias, relu);
  }
}

extern "C" void kernel_launch(void* const* d_in, const int* in_sizes, int n_in,
                              void* d_out, int out_size, void* d_ws, size_t ws_size,
                              hipStream_t stream) {
  (void)in_sizes; (void)n_in; (void)out_size; (void)ws_size;
  const float* x     = (const float*)d_in[0];
  const float* y     = (const float*)d_in[1];
  // d_in[2] = y_mask (exact causal tril; handled analytically)
  const float* Wq    = (const float*)d_in[3];
  const float* bq    = (const float*)d_in[4];
  const float* Wk    = (const float*)d_in[5];
  const float* bk    = (const float*)d_in[6];
  const float* Wv    = (const float*)d_in[7];
  const float* bv    = (const float*)d_in[8];
  const float* W1    = (const float*)d_in[9];
  const float* b1    = (const float*)d_in[10];
  const float* ln1g  = (const float*)d_in[11];
  const float* ln1b  = (const float*)d_in[12];
  const float* W2    = (const float*)d_in[13];
  const float* b2    = (const float*)d_in[14];
  const float* ln2g  = (const float*)d_in[15];
  const float* ln2b  = (const float*)d_in[16];
  const float* Wf1   = (const float*)d_in[17];
  const float* bf1   = (const float*)d_in[18];
  const float* Wf2   = (const float*)d_in[19];
  const float* bf2   = (const float*)d_in[20];
  const float* ln3g  = (const float*)d_in[21];
  const float* ln3b  = (const float*)d_in[22];
  float* out = (float*)d_out;

  // ---- workspace carve-out (~205 MB) ----
  char* ws = (char*)d_ws;
  size_t off = 0;
  auto carve = [&](size_t bytes) -> char* {
    char* p = ws + off;
    off += (bytes + 255) & ~(size_t)255;
    return p;
  };
  u16*   ybf   = (u16*)  carve((size_t)BS_ * D_ * 2);       // y bf16
  u16*   xbf   = (u16*)  carve((size_t)BS_ * D_ * 2);       // x bf16
  u16*   wqkv  = (u16*)  carve((size_t)D_ * 3 * D_ * 2);    // packed QKV weights
  u16*   w1b   = (u16*)  carve((size_t)D_ * D_ * 2);
  u16*   w2s   = (u16*)  carve((size_t)D_ * D_ * 2);        // head-summed W2
  u16*   wf1b  = (u16*)  carve((size_t)D_ * DF_ * 2);
  u16*   wf2b  = (u16*)  carve((size_t)DF_ * D_ * 2);
  float* bqkv  = (float*)carve((size_t)3 * D_ * 4);         // packed QKV bias
  u16*   qkvbf = (u16*)  carve((size_t)BS_ * 3 * D_ * 2);   // fused QKV activations
  float* scf   = (float*)carve((size_t)B_ * S_ * S_ * 4);   // scores fp32 (per head / cross)
  u16*   pbf   = (u16*)  carve((size_t)B_ * S_ * S_ * 2);   // probs bf16
  u16*   catbf = (u16*)  carve((size_t)BS_ * D_ * 2);       // self-attn concat
  u16*   a1bf  = (u16*)  carve((size_t)BS_ * D_ * 2);
  u16*   a2bf  = (u16*)  carve((size_t)BS_ * D_ * 2);
  u16*   at2bf = (u16*)  carve((size_t)BS_ * D_ * 2);       // cross-attn out
  u16*   h1bf  = (u16*)  carve((size_t)BS_ * DF_ * 2);      // FFN hidden
  float* lnf   = (float*)carve((size_t)BS_ * D_ * 4);       // pre-LN fp32

  const float scale = 0.125f;  // 1/sqrt(64)
  dim3 blk(256);

  // ---- precision prep ----
  k_cvt_bf16<<<(BS_*D_ + 255)/256, blk, 0, stream>>>(y, ybf, BS_*D_);
  k_cvt_bf16<<<(BS_*D_ + 255)/256, blk, 0, stream>>>(x, xbf, BS_*D_);
  k_pack_qkvw<<<(D_*3*D_ + 255)/256, blk, 0, stream>>>(Wq, Wk, Wv, wqkv);
  k_pack_qkvb<<<(3*D_ + 255)/256, blk, 0, stream>>>(bq, bk, bv, bqkv);
  k_cvt_bf16<<<(D_*D_ + 255)/256, blk, 0, stream>>>(W1, w1b, D_*D_);
  k_sum_w2<<<(D_*D_ + 255)/256, blk, 0, stream>>>(W2, w2s);
  k_cvt_bf16<<<(D_*DF_ + 255)/256, blk, 0, stream>>>(Wf1, wf1b, D_*DF_);
  k_cvt_bf16<<<(DF_*D_ + 255)/256, blk, 0, stream>>>(Wf2, wf2b, DF_*D_);

  // ---- fused QKV projection: [BS,D] x [D,3D] -> bf16 (+bias) ----
  launch_gemm(0, ybf, 0, D_, wqkv, 0, 3*D_, nullptr, qkvbf, 0, 3*D_,
              BS_, 3*D_, D_, 1.f, bqkv, 0, 1, stream);

  // ---- masked multi-head self attention (loop heads, batched over B) ----
  for (int h = 0; h < H_; ++h) {
    const u16* qh = qkvbf + h * DK_;              // [b][s][h*64+kk], ld 3D
    const u16* kh = qkvbf + D_ + h * DK_;
    const u16* vh = qkvbf + 2 * D_ + h * DK_;
    // scores = Q Kt * scale  : NT GEMM, K=64
    launch_gemm(1, qh, (long long)S_*3*D_, 3*D_, kh, (long long)S_*3*D_, 3*D_,
                scf, nullptr, (long long)S_*S_, S_,
                S_, S_, DK_, scale, nullptr, 0, B_, stream);
    k_softmax_rows<<<B_*S_, blk, 0, stream>>>(scf, pbf, 1);
    // attn = P V -> cat[:, h*64:]
    launch_gemm(0, pbf, (long long)S_*S_, S_, vh, (long long)S_*3*D_, 3*D_,
                nullptr, catbf + h * DK_, (long long)S_*D_, D_,
                S_, DK_, S_, 1.f, nullptr, 0, B_, stream);
  }

  // ---- a1 = LN(y + cat @ W1 + b1) ----
  launch_gemm(0, catbf, 0, D_, w1b, 0, D_, lnf, nullptr, 0, D_,
              BS_, D_, D_, 1.f, nullptr, 0, 1, stream);
  k_res_bias_ln<<<BS_, blk, 0, stream>>>(lnf, y, b1, ln1g, ln1b, a1bf, nullptr);

  // ---- cross attention: sc2 = a1 x^T * scale ; p2 ; attn2 = p2 x ----
  launch_gemm(1, a1bf, (long long)S_*D_, D_, xbf, (long long)S_*D_, D_,
              scf, nullptr, (long long)S_*S_, S_,
              S_, S_, D_, scale, nullptr, 0, B_, stream);
  k_softmax_rows<<<B_*S_, blk, 0, stream>>>(scf, pbf, 0);
  launch_gemm(0, pbf, (long long)S_*S_, S_, xbf, (long long)S_*D_, D_,
              nullptr, at2bf, (long long)S_*D_, D_,
              S_, D_, S_, 1.f, nullptr, 0, B_, stream);

  // ---- a2 = LN(y + attn2 @ sum_h(W2) + b2)  [tile(attn2,h)@W2 collapsed] ----
  launch_gemm(0, at2bf, 0, D_, w2s, 0, D_, lnf, nullptr, 0, D_,
              BS_, D_, D_, 1.f, nullptr, 0, 1, stream);
  k_res_bias_ln<<<BS_, blk, 0, stream>>>(lnf, y, b2, ln2g, ln2b, a2bf, nullptr);

  // ---- FFN: h1 = relu(a2 @ Wf1 + bf1) (bf16 fused) ; ff2 = h1 @ Wf2 ----
  launch_gemm(0, a2bf, 0, D_, wf1b, 0, DF_, nullptr, h1bf, 0, DF_,
              BS_, DF_, D_, 1.f, bf1, 1, 1, stream);
  launch_gemm(0, h1bf, 0, DF_, wf2b, 0, D_, lnf, nullptr, 0, D_,
              BS_, D_, DF_, 1.f, nullptr, 0, 1, stream);

  // ---- out = LN(y + ff + bf2) (fp32) ----
  k_res_bias_ln<<<BS_, blk, 0, stream>>>(lnf, y, bf2, ln3g, ln3b, nullptr, out);
}
